// SplatAwareFeedForward_38431367364866
// MI455X (gfx1250) — matrix operand order repaired and checked by reading, not compiled, not run on hardware
//
#include <hip/hip_runtime.h>
#include <hip/hip_bf16.h>

typedef unsigned short u16;
typedef __attribute__((ext_vector_type(16))) __bf16 v16bf;
typedef __attribute__((ext_vector_type(8)))  float  v8f;

union FragU { uint4 q[2]; v16bf v; };

__device__ __forceinline__ u16 f2bf(float f) {
  unsigned u = __float_as_uint(f);
  unsigned r = 0x7FFFu + ((u >> 16) & 1u);
  return (u16)((u + r) >> 16);
}

__device__ __forceinline__ float gelu_exact(float v) {
  return 0.5f * v * (1.0f + erff(v * 0.70710678118654752f));
}

// A fragment: 16x32 bf16, row-major [row][K] source, per ISA 16-bit A layout.
// lane&15 = M row; half=lane>>4: elems 0..7 = K kb+half*8.., elems 8..15 = K kb+16+half*8..
__device__ __forceinline__ v16bf load_a(const u16* row_base, int kb, int half) {
  FragU f;
  f.q[0] = *(const uint4*)(row_base + kb + half * 8);
  f.q[1] = *(const uint4*)(row_base + kb + 16 + half * 8);
  return f.v;
}

// B fragment: 32x16 bf16 from pre-transposed weights Wt[N][K] row-major.
// lane&15 = N col; half=lane>>4: elems 0..7 = K kb+half*8.., elems 8..15 = K kb+16+half*8..
__device__ __forceinline__ v16bf load_b(const u16* row_base, int kb, int half) {
  FragU f;
  f.q[0] = *(const uint4*)(row_base + kb + half * 8);
  f.q[1] = *(const uint4*)(row_base + kb + 16 + half * 8);
  return f.v;
}

#define WMMA_BF16(a, b, c) \
  __builtin_amdgcn_wmma_f32_16x16x32_bf16(false, (a), false, (b), (short)0, (c), false, false)

// ---------------------------------------------------------------------------
// Transpose + fp32->bf16 convert: dst[slab][c][r] = src[slab][r][c]
// ---------------------------------------------------------------------------
__global__ void tconv_kernel(const float* __restrict__ src, u16* __restrict__ dst,
                             int slabs, int R, int C) {
  size_t total = (size_t)slabs * R * C;
  size_t i = (size_t)blockIdx.x * 256 + threadIdx.x;
  if (i >= total) return;
  size_t rc = (size_t)R * C;
  size_t slab = i / rc;
  size_t rem = i - slab * rc;
  int c = (int)(rem / R);
  int r = (int)(rem - (size_t)c * R);
  dst[i] = f2bf(src[slab * rc + (size_t)r * C + c]);
}

// ---------------------------------------------------------------------------
// Bucket sequence positions by expert; emit tiles of up to 4 positions.
// ---------------------------------------------------------------------------
__global__ void bucket_kernel(const int* __restrict__ splat_id,
                              int* __restrict__ sorted_s,
                              int4* __restrict__ tiles,
                              int* __restrict__ meta) {
  __shared__ int cnt[16];
  __shared__ int base[16];
  __shared__ int fill[16];
  int tid = threadIdx.x;
  if (tid < 16) { cnt[tid] = 0; fill[tid] = 0; }
  __syncthreads();
  for (int s = tid; s < 4096; s += 256) atomicAdd(&cnt[splat_id[s]], 1);
  __syncthreads();
  if (tid == 0) {
    int off = 0, t = 0;
    for (int e = 0; e < 16; ++e) {
      base[e] = off;
      int c = cnt[e];
      for (int i = 0; i < c; i += 4) {
        int4 ti; ti.x = e; ti.y = off + i; ti.z = (c - i < 4) ? (c - i) : 4; ti.w = 0;
        tiles[t++] = ti;
      }
      off += c;
    }
    meta[0] = t;
  }
  __syncthreads();
  for (int s = tid; s < 4096; s += 256) {
    int e = splat_id[s];
    int pos = base[e] + atomicAdd(&fill[e], 1);
    sorted_s[pos] = s;
  }
}

// ---------------------------------------------------------------------------
// Expert MLP: per tile (16 token rows, one expert):
//   H = gelu(X @ W1[e] + b1[e]);  routed = H @ W2[e] + b2[e]  (bf16 out)
// ---------------------------------------------------------------------------
__global__ __launch_bounds__(256) void expert_kernel(
    const float* __restrict__ x,
    const u16* __restrict__ W1t,   // [E][H=256][D=1024] bf16
    const float* __restrict__ b1,  // [E][256]
    const u16* __restrict__ W2t,   // [E][D=1024][H=256] bf16
    const float* __restrict__ b2,  // [E][1024]
    const int* __restrict__ sorted_s,
    const int4* __restrict__ tiles,
    const int* __restrict__ meta,
    u16* __restrict__ routed)      // [B*S][1024] bf16
{
  if ((int)blockIdx.x >= meta[0]) return;
  int4 ti = tiles[blockIdx.x];
  const int e = ti.x, sbase = ti.y, scount = ti.z;

  __shared__ __align__(16) u16 Xs[16 * 1024];   // X tile bf16; reused as output stage
  __shared__ __align__(16) u16 Hs[16 * 256];    // hidden tile bf16
  __shared__ int s_ids[4];

  const int tid = threadIdx.x;
  if (tid < 4) {
    int i = (tid < scount) ? tid : (scount - 1);  // pad with duplicate (benign rewrite)
    s_ids[tid] = sorted_s[sbase + i];
  }
  __syncthreads();

  // Stage X tile: row m -> (batch = m&3, seq = s_ids[m>>2]); fp32 -> bf16
  {
    int m = tid >> 4, lane16 = tid & 15;
    int b = m & 3, s = s_ids[m >> 2];
    const float* xr = x + ((size_t)b * 4096 + s) * 1024;
    for (int j = 0; j < 16; ++j) {
      int col = lane16 * 64 + j * 4;
      float4 f = *(const float4*)(xr + col);
      unsigned p0 = (unsigned)f2bf(f.x) | ((unsigned)f2bf(f.y) << 16);
      unsigned p1 = (unsigned)f2bf(f.z) | ((unsigned)f2bf(f.w) << 16);
      unsigned* d = (unsigned*)(Xs + m * 1024 + col);
      d[0] = p0; d[1] = p1;
    }
  }
  __syncthreads();

  const int wave = tid >> 5, lane = tid & 31;
  const int half = lane >> 4, l15 = lane & 15;
  const u16* arow = Xs + l15 * 1024;

  // Layer 1: M=16, N=256, K=1024. Each wave: 2 N-tiles sharing the A fragment.
  {
    const u16* W1e = W1t + (size_t)e * 256 * 1024;
    int n0 = wave * 32;
    const u16* brow0 = W1e + (size_t)(n0 + l15) * 1024;
    const u16* brow1 = W1e + (size_t)(n0 + 16 + l15) * 1024;
    v8f acc0 = {}; v8f acc1 = {};
    for (int kb = 0; kb < 1024; kb += 32) {
      v16bf a  = load_a(arow, kb, half);
      v16bf bA = load_b(brow0, kb, half);
      v16bf bB = load_b(brow1, kb, half);
      acc0 = WMMA_BF16(a, bA, acc0);
      acc1 = WMMA_BF16(a, bB, acc1);
    }
    float bias0 = b1[e * 256 + n0 + l15];
    float bias1 = b1[e * 256 + n0 + 16 + l15];
    for (int r = 0; r < 8; ++r) {
      int mrow = half * 8 + r;
      Hs[mrow * 256 + n0 + l15]      = f2bf(gelu_exact(acc0[r] + bias0));
      Hs[mrow * 256 + n0 + 16 + l15] = f2bf(gelu_exact(acc1[r] + bias1));
    }
  }
  __syncthreads();

  // Layer 2: M=16, N=1024, K=256. Each wave: 8 N-tiles. Output staged in Xs.
  {
    const u16* W2e = W2t + (size_t)e * 1024 * 256;
    const u16* hrow = Hs + l15 * 256;
    u16* Os = Xs;
    for (int t = 0; t < 8; ++t) {
      int n0 = (wave * 8 + t) * 16;
      const u16* brow = W2e + (size_t)(n0 + l15) * 256;
      v8f acc = {};
      for (int kb = 0; kb < 256; kb += 32) {
        v16bf a = load_a(hrow, kb, half);
        v16bf b = load_b(brow, kb, half);
        acc = WMMA_BF16(a, b, acc);
      }
      float bias = b2[e * 1024 + n0 + l15];
      for (int r = 0; r < 8; ++r)
        Os[(half * 8 + r) * 1024 + n0 + l15] = f2bf(acc[r] + bias);
    }
  }
  __syncthreads();

  // Coalesced bf16 store of the routed tile.
  {
    int m = tid >> 4, lane16 = tid & 15;
    int b = m & 3, s = s_ids[m >> 2];
    u16* rr = routed + ((size_t)b * 4096 + s) * 1024;
    for (int j = 0; j < 8; ++j) {
      int col = lane16 * 64 + j * 8;
      *(uint4*)(rr + col) = *(const uint4*)(Xs + m * 1024 + col);
    }
  }
}

// ---------------------------------------------------------------------------
// Aggregator GEMM: out = (gelu?)(A[16384x1024] @ Bt^T + bias)
// Bt is pre-transposed [N=1024][K=1024] bf16. One block per 32 rows: each
// wave keeps two accumulators per N-tile so every B fragment feeds 2 WMMAs
// (halves L2 weight traffic vs a 16-row M-tile).
// ---------------------------------------------------------------------------
__global__ __launch_bounds__(256) void agg_kernel(
    const u16* __restrict__ Abf, const u16* __restrict__ Bt,
    const float* __restrict__ bias,
    u16* __restrict__ outBf, float* __restrict__ outF, int do_gelu)
{
  __shared__ __align__(16) u16 As[32 * 1024];   // 64 KB
  const int tid = threadIdx.x;
  const size_t row0 = (size_t)blockIdx.x * 32;

  // Stage 32 A rows (bf16, contiguous): 8 threads per row, 16 uint4 each.
  {
    int m = tid >> 3, lane8 = tid & 7;
    const u16* src = Abf + (row0 + m) * 1024;
    for (int j = 0; j < 16; ++j) {
      int col = lane8 * 128 + j * 8;
      *(uint4*)(As + m * 1024 + col) = *(const uint4*)(src + col);
    }
  }
  __syncthreads();

  const int wave = tid >> 5, lane = tid & 31;
  const int half = lane >> 4, l15 = lane & 15;
  const u16* arow0 = As + l15 * 1024;          // rows 0..15
  const u16* arow1 = As + (16 + l15) * 1024;   // rows 16..31

  for (int t = 0; t < 8; ++t) {
    int n0 = (wave * 8 + t) * 16;
    const u16* brow = Bt + (size_t)(n0 + l15) * 1024;
    v8f acc0 = {}; v8f acc1 = {};
    for (int kb = 0; kb < 1024; kb += 32) {
      v16bf b  = load_b(brow, kb, half);
      v16bf a0 = load_a(arow0, kb, half);
      v16bf a1 = load_a(arow1, kb, half);
      acc0 = WMMA_BF16(a0, b, acc0);
      acc1 = WMMA_BF16(a1, b, acc1);
    }
    float bb = bias[n0 + l15];
    for (int r = 0; r < 8; ++r) {
      float v0 = acc0[r] + bb;
      float v1 = acc1[r] + bb;
      if (do_gelu) { v0 = gelu_exact(v0); v1 = gelu_exact(v1); }
      size_t idx0 = (row0 + half * 8 + r) * 1024 + n0 + l15;
      size_t idx1 = (row0 + 16 + half * 8 + r) * 1024 + n0 + l15;
      if (outBf) { outBf[idx0] = f2bf(v0); outBf[idx1] = f2bf(v1); }
      else       { outF[idx0] = v0;        outF[idx1] = v1; }
    }
  }
}

// ---------------------------------------------------------------------------
extern "C" void kernel_launch(void* const* d_in, const int* in_sizes, int n_in,
                              void* d_out, int out_size, void* d_ws, size_t ws_size,
                              hipStream_t stream) {
  const float* x    = (const float*)d_in[0];
  const int*   sid  = (const int*)d_in[1];
  const float* W1   = (const float*)d_in[2];
  const float* b1   = (const float*)d_in[3];
  const float* W2   = (const float*)d_in[4];
  const float* b2   = (const float*)d_in[5];
  const float* Wg1  = (const float*)d_in[6];
  const float* bg1  = (const float*)d_in[7];
  const float* Wg2  = (const float*)d_in[8];
  const float* bg2  = (const float*)d_in[9];
  float* out = (float*)d_out;

  char* ws = (char*)d_ws;
  int*  sorted_s = (int*)(ws + 0);                 // 16 KB
  int4* tiles    = (int4*)(ws + (64 << 10));       // ~17 KB
  int*  meta     = (int*)(ws + (96 << 10));        // 4 B
  u16*  W1t      = (u16*)(ws + (size_t)(1)  * (1 << 20));  // 8 MB
  u16*  W2t      = (u16*)(ws + (size_t)(9)  * (1 << 20));  // 8 MB
  u16*  Wg1t     = (u16*)(ws + (size_t)(17) * (1 << 20));  // 2 MB
  u16*  Wg2t     = (u16*)(ws + (size_t)(19) * (1 << 20));  // 2 MB
  u16*  routed   = (u16*)(ws + (size_t)(21) * (1 << 20));  // 32 MB
  u16*  g        = (u16*)(ws + (size_t)(53) * (1 << 20));  // 32 MB

  // Weight transpose + bf16 convert
  tconv_kernel<<<(16 * 1024 * 256 + 255) / 256, 256, 0, stream>>>(W1, W1t, 16, 1024, 256);
  tconv_kernel<<<(16 * 256 * 1024 + 255) / 256, 256, 0, stream>>>(W2, W2t, 16, 256, 1024);
  tconv_kernel<<<(1024 * 1024 + 255) / 256, 256, 0, stream>>>(Wg1, Wg1t, 1, 1024, 1024);
  tconv_kernel<<<(1024 * 1024 + 255) / 256, 256, 0, stream>>>(Wg2, Wg2t, 1, 1024, 1024);

  // Expert routing
  bucket_kernel<<<1, 256, 0, stream>>>(sid, sorted_s, tiles, meta);

  // Expert MLP (max tiles = 4096/4 + 16 = 1040)
  expert_kernel<<<1040, 256, 0, stream>>>(x, W1t, b1, W2t, b2, sorted_s, tiles, meta, routed);

  // Aggregator MLP (32-row M tiles -> 512 blocks)
  agg_kernel<<<512, 256, 0, stream>>>(routed, Wg1t, bg1, g, nullptr, 1);
  agg_kernel<<<512, 256, 0, stream>>>(g, Wg2t, bg2, nullptr, out, 0);
}
